// CrossAttn_75033078661891
// MI455X (gfx1250) — matrix operand order repaired and checked by reading, not compiled, not run on hardware
//
#include <hip/hip_runtime.h>
#include <hip/hip_bf16.h>

// ---------------------------------------------------------------------------
// CDNA5 (gfx1250) cross-attention: Q/KV projections + fused RMSNorm,
// flash-attention, output projection. All matmuls on v_wmma_f32_16x16x32_f16.
// Softmax runs in base-2 with 0.125*log2(e) folded into the stored q, and a
// STATIC shift derived from the RMSNorm row-norm bound (exact by softmax
// shift-invariance) -> no cross-lane max reductions in the inner loop.
// V is produced pre-transposed ([B,H,HD,Sc]) so attention stage-in is a pure
// b128 copy for both K and V.
// ---------------------------------------------------------------------------

typedef __attribute__((ext_vector_type(16))) _Float16 v16h;
typedef __attribute__((ext_vector_type(8)))  _Float16 v8h;
typedef __attribute__((ext_vector_type(4)))  _Float16 v4h;
typedef __attribute__((ext_vector_type(8)))  float    v8f;

#define ATTN_H   16
#define ATTN_HD  64

__device__ __forceinline__ v8f vzero8() {
  v8f z = {0.f, 0.f, 0.f, 0.f, 0.f, 0.f, 0.f, 0.f};
  return z;
}

__device__ __forceinline__ v8f wmma_f16(v16h a, v16h b, v8f c) {
  // D = A(16x32 f16) * B(32x16 f16) + C(16x16 f32)
  return __builtin_amdgcn_wmma_f32_16x16x32_f16(
      /*neg_a=*/false, a, /*neg_b=*/false, b,
      /*c_mod=*/(short)0, c, /*reuse_a=*/false, /*reuse_b=*/false);
}

// Raw v_exp_f32: inputs here are in [-inf, 0], results >= 2^-40 (normal f32),
// so the libm denormal fixup (cmp/cndmask/ldexp) is unnecessary.
__device__ __forceinline__ float fast_exp2(float x) {
  return __builtin_amdgcn_exp2f(x);
}

// 4-element vector load + convert to f16 (float4 -> b128 global load).
__device__ __forceinline__ v4h ld4h(const float* p) {
  const float4 v = *reinterpret_cast<const float4*>(p);
  v4h h;
  h[0] = (_Float16)v.x; h[1] = (_Float16)v.y;
  h[2] = (_Float16)v.z; h[3] = (_Float16)v.w;
  return h;
}
__device__ __forceinline__ v4h ld4h(const _Float16* p) {
  return *reinterpret_cast<const v4h*>(p);
}

// A fragment: 16x32 (MxK) tile, row-major source, row stride `ld` (halfs).
// ISA: lane l holds M = l&15; lanes<16 K-runs {0..7,16..23}, lanes>=16
// K-runs {8..15,24..31}. Two contiguous 8-half runs -> ds_load_b128 friendly.
__device__ __forceinline__ v16h load_a(const _Float16* p, int ld) {
  const int lane = threadIdx.x & 31;
  const int m    = lane & 15;
  const int kb   = (lane < 16) ? 0 : 8;
  const _Float16* row = p + (size_t)m * ld;
  v16h a;
#pragma unroll
  for (int j = 0; j < 8; ++j) a[j] = row[kb + j];
#pragma unroll
  for (int j = 0; j < 8; ++j) a[8 + j] = row[16 + kb + j];
  return a;
}

// B fragment: 32x16 (KxN) tile from a TRANSPOSED (N-major: p[n*ld + k]) source.
// ISA: lane l holds N = l&15; lanes<16 K-runs {0..7,16..23}, lanes>=16
// K-runs {8..15,24..31} (same K order as load_a).
__device__ __forceinline__ v16h load_bt(const _Float16* p, int ld) {
  const int lane = threadIdx.x & 31;
  const int n    = lane & 15;
  const int kb   = (lane < 16) ? 0 : 8;
  const _Float16* row = p + (size_t)n * ld;
  v16h b;
#pragma unroll
  for (int j = 0; j < 8; ++j) b[j] = row[kb + j];
#pragma unroll
  for (int j = 0; j < 8; ++j) b[8 + j] = row[16 + kb + j];
  return b;
}

// ---------------------------------------------------------------------------
// LDS-staged WMMA GEMM with fused epilogues.
//   MODE_Q   : C = A@B + bias, per-64-col-head RMSNorm(gamma*qscale), f16 out
//   MODE_KV  : cols [0,1024) -> k (RMSNorm(gamma), [B,H,Sc,HD] f16),
//              cols [1024,2048) -> v ([B,H,HD,Sc] f16, PRE-TRANSPOSED)
//   MODE_PROJ: C = A@B + bias, f32 row-major out
// Block: 256 threads (8 waves), 128(M) x 128(N) tile, K-step 64.
// Wave (mw = w>>1, nw = w&1) owns a 32(M) x 64(N) subtile: 2x4 C fragments.
// ---------------------------------------------------------------------------
constexpr int MODE_Q    = 0;
constexpr int MODE_KV   = 1;
constexpr int MODE_PROJ = 2;

template <int MODE, typename TA>
__global__ __launch_bounds__(256, 1)
void gemm_epi_kernel(const TA* __restrict__ A,
                     const float* __restrict__ Bw,
                     const float* __restrict__ bias,
                     const float* __restrict__ gamma,
                     _Float16* __restrict__ outK,
                     _Float16* __restrict__ outV,
                     float* __restrict__ outF,
                     int M, int N, int K, int Sc, float qscale) {
  __shared__ _Float16 lA[128][72];    // 128x64 A tile, row-major (padded)
  __shared__ _Float16 lBt[128][72];   // 64x128 B tile stored TRANSPOSED [n][k]

  const int tid  = threadIdx.x;
  const int w    = tid >> 5;
  const int mw   = w >> 1;            // 0..3 : 32-row group
  const int nw   = w & 1;             // 0..1 : 64-col group
  const int lane = tid & 31;
  const int half = lane >> 4;
  const int nl   = lane & 15;
  const int n0   = blockIdx.x * 128;
  const int row0 = blockIdx.y * 128;

  v8f acc[2][4];
#pragma unroll
  for (int i = 0; i < 2; ++i)
#pragma unroll
    for (int t = 0; t < 4; ++t) acc[i][t] = vzero8();

  for (int k0 = 0; k0 < K; k0 += 64) {
    // Stage A (-> f16, row-major) and B (-> f16, transposed) tiles into LDS.
#pragma unroll
    for (int i = 0; i < 8; ++i) {
      int idx = tid + 256 * i;
      int r = idx >> 4, c4 = idx & 15;
      v4h hv = ld4h(&A[(size_t)(row0 + r) * K + (k0 + c4 * 4)]);
      *reinterpret_cast<v4h*>(&lA[r][c4 * 4]) = hv;
    }
#pragma unroll
    for (int i = 0; i < 8; ++i) {
      int idx = tid + 256 * i;
      int r = idx >> 5, c4 = idx & 31;          // r: k 0..63, c4: n-chunk
      v4h hv = ld4h(&Bw[(size_t)(k0 + r) * N + (n0 + c4 * 4)]);
#pragma unroll
      for (int j = 0; j < 4; ++j) lBt[c4 * 4 + j][r] = hv[j];
    }
    __syncthreads();

    v16h af[2][2];
#pragma unroll
    for (int i = 0; i < 2; ++i)
#pragma unroll
      for (int s = 0; s < 2; ++s)
        af[i][s] = load_a(&lA[32 * mw + 16 * i][32 * s], 72);
#pragma unroll
    for (int t = 0; t < 4; ++t) {
#pragma unroll
      for (int s = 0; s < 2; ++s) {
        v16h bf = load_bt(&lBt[64 * nw + 16 * t][32 * s], 72);
#pragma unroll
        for (int i = 0; i < 2; ++i) acc[i][t] = wmma_f16(af[i][s], bf, acc[i][t]);
      }
    }
    __syncthreads();
  }

  const int cb = n0 + 64 * nw;        // this wave's 64-col (one head) base

  // ----- epilogue: bias -----
#pragma unroll
  for (int t = 0; t < 4; ++t) {
    float bv = bias[cb + 16 * t + nl];
#pragma unroll
    for (int i = 0; i < 2; ++i)
#pragma unroll
      for (int r = 0; r < 8; ++r) acc[i][t][r] += bv;
  }

  if (MODE == MODE_PROJ) {
#pragma unroll
    for (int i = 0; i < 2; ++i)
#pragma unroll
      for (int t = 0; t < 4; ++t)
#pragma unroll
        for (int r = 0; r < 8; ++r) {
          int grow = row0 + 32 * mw + 16 * i + r + 8 * half;
          outF[(size_t)grow * N + (cb + 16 * t + nl)] = acc[i][t][r];
        }
    return;
  }

  // ----- per-head RMSNorm (wave's 64-col group == one head) -----
  const bool is_k_part = (MODE == MODE_Q) || (cb < (N >> 1));
  if (is_k_part) {
    float ss[2][8];
#pragma unroll
    for (int i = 0; i < 2; ++i)
#pragma unroll
      for (int r = 0; r < 8; ++r) {
        float s = 0.f;
#pragma unroll
        for (int t = 0; t < 4; ++t) s += acc[i][t][r] * acc[i][t][r];
        ss[i][r] = s;
      }
    for (int mask = 1; mask < 16; mask <<= 1) {
      float tmp[2][8];
#pragma unroll
      for (int i = 0; i < 2; ++i)
#pragma unroll
        for (int r = 0; r < 8; ++r) tmp[i][r] = __shfl_xor(ss[i][r], mask, 32);
#pragma unroll
      for (int i = 0; i < 2; ++i)
#pragma unroll
        for (int r = 0; r < 8; ++r) ss[i][r] += tmp[i][r];
    }
#pragma unroll
    for (int i = 0; i < 2; ++i)
#pragma unroll
      for (int r = 0; r < 8; ++r) {
        float inv = rsqrtf(ss[i][r] * (1.f / 64.f) + 1e-6f);
#pragma unroll
        for (int t = 0; t < 4; ++t) acc[i][t][r] *= inv;
      }
#pragma unroll
    for (int t = 0; t < 4; ++t) {
      // qscale folds 1/sqrt(hd) * log2(e) into stored q (1.0 for k).
      float g = gamma[16 * t + nl] * qscale;
#pragma unroll
      for (int i = 0; i < 2; ++i)
#pragma unroll
        for (int r = 0; r < 8; ++r) acc[i][t][r] *= g;
    }
  }

#pragma unroll
  for (int i = 0; i < 2; ++i)
#pragma unroll
    for (int t = 0; t < 4; ++t)
#pragma unroll
      for (int r = 0; r < 8; ++r) {
        int grow = row0 + 32 * mw + 16 * i + r + 8 * half;
        int hd   = 16 * t + nl;
        _Float16 hv = (_Float16)acc[i][t][r];
        if (MODE == MODE_Q) {
          outK[(size_t)grow * N + (cb + hd)] = hv;               // [B*S, D]
        } else {                                                 // MODE_KV
          int hh   = (cb & 1023) >> 6;
          int bb   = grow / Sc;
          int trow = grow % Sc;
          if (is_k_part) {                                       // [B,H,Sc,HD]
            outK[(((size_t)(bb * ATTN_H + hh) * Sc) + trow) * ATTN_HD + hd] = hv;
          } else {                                               // [B,H,HD,Sc]
            outV[(((size_t)(bb * ATTN_H + hh) * ATTN_HD) + hd) * Sc + trow] = hv;
          }
        }
      }
}

// ---------------------------------------------------------------------------
// Rigorous log2-domain score bound from the RMSNorm gammas:
//   |score_log2| <= ||q||*||k|| * (0.125*log2e) = 8*log2e * max|gq| * max|gk|
// (+0.5 margin for f16 rounding of q/k). Computed on device: graph-safe.
// ---------------------------------------------------------------------------
__global__ void score_bound_kernel(const float* __restrict__ gq,
                                   const float* __restrict__ gk,
                                   float* __restrict__ bound) {
  if (threadIdx.x == 0) {
    float mq = 0.f, mk = 0.f;
    for (int i = 0; i < ATTN_HD; ++i) {
      mq = fmaxf(mq, fabsf(gq[i]));
      mk = fmaxf(mk, fabsf(gk[i]));
    }
    *bound = 8.f * 1.44269504088896340736f * mq * mk + 0.5f;
  }
}

// ---------------------------------------------------------------------------
// Flash attention, static-shift softmax. One wave per 16-row query tile;
// 4 waves share LDS-staged 128x64 K and 64x128 V^T tiles (both b128 copies).
// Per 128-ctx-col iteration: 16 score WMMAs -> raw v_exp_f32 -> P to LDS ->
// 4 rowsum WMMAs (P @ ones) -> 16 PV WMMAs. No cross-lane ops in the loop.
// grid = (S/64, H, B), block = 128.
// ---------------------------------------------------------------------------
__global__ __launch_bounds__(128, 1)
void flash_attn_kernel(const _Float16* __restrict__ qh,
                       const _Float16* __restrict__ kh,
                       const _Float16* __restrict__ vh,
                       _Float16* __restrict__ oh,
                       const float* __restrict__ bound) {
  constexpr int S = 2048, Sc = 2048, D = 1024;
  __shared__ _Float16 lK[128][72];     // [t][hd]   (load_bt: B[k=hd][n=t])
  __shared__ _Float16 lVt[64][136];    // [hd][t]   (load_bt: B[k=t][n=hd])
  __shared__ _Float16 lP[4][16][136];  // per-wave 16x128 P staging [m][t]

  const int tid  = threadIdx.x;
  const int w    = tid >> 5;
  const int lane = tid & 31;
  const int half = lane >> 4;
  const int nl   = lane & 15;
  const int b    = blockIdx.z;
  const int h    = blockIdx.y;
  const int s0   = blockIdx.x * 64 + 16 * w;

  const float mshift = *bound;   // uniform scalar load

  // Q fragments for this wave's 16 rows, K-dim = HD = 64 (two k-steps).
  const _Float16* qbase = qh + (size_t)(b * S + s0) * D + h * ATTN_HD;
  const v16h qa0 = load_a(qbase, D);
  const v16h qa1 = load_a(qbase + 32, D);

  v16h ones;
#pragma unroll
  for (int j = 0; j < 16; ++j) ones[j] = (_Float16)1.f;

  v8f acc[4];
#pragma unroll
  for (int t = 0; t < 4; ++t) acc[t] = vzero8();
  float lrow[8];
#pragma unroll
  for (int r = 0; r < 8; ++r) lrow[r] = 0.f;

  const _Float16* kbase = kh + (size_t)((b * ATTN_H + h) * Sc) * ATTN_HD;
  const _Float16* vbase = vh + (size_t)((b * ATTN_H + h) * ATTN_HD) * Sc;
  const v8h* ksrc = reinterpret_cast<const v8h*>(kbase);
  const v8h* vsrc = reinterpret_cast<const v8h*>(vbase);

  for (int t0 = 0; t0 < Sc; t0 += 128) {
    // Stage K (128 t-rows x 64 hd) and V^T (64 hd-rows x 128 t): pure b128
    // copies, 1024 8-half chunks each, 8 per thread per array.
#pragma unroll
    for (int i = 0; i < 8; ++i) {
      int idx = tid + 128 * i;
      int kr = idx >> 3, kc = idx & 7;
      *reinterpret_cast<v8h*>(&lK[kr][kc * 8]) =
          ksrc[(size_t)(t0 + kr) * 8 + kc];
      int vr = idx >> 4, vc = idx & 15;
      *reinterpret_cast<v8h*>(&lVt[vr][vc * 8]) =
          vsrc[(size_t)vr * (Sc / 8) + (t0 >> 3) + vc];
    }
    if (t0 + 128 < Sc) {
      __builtin_prefetch(kbase + (size_t)(t0 + 128) * ATTN_HD, 0, 1);
      __builtin_prefetch(vbase + (size_t)(t0 + 128), 0, 1);
    }
    __syncthreads();

    // Scores (pre-scaled, log2 domain): process tile-pairs with all four B
    // fragments loaded up front for load/WMMA overlap, then
    // P = exp2(s - static_shift) straight into the per-wave LDS A-tile.
#pragma unroll
    for (int tp = 0; tp < 4; ++tp) {
      v16h b00 = load_bt(&lK[(2 * tp) * 16][0], 72);
      v16h b01 = load_bt(&lK[(2 * tp) * 16][32], 72);
      v16h b10 = load_bt(&lK[(2 * tp + 1) * 16][0], 72);
      v16h b11 = load_bt(&lK[(2 * tp + 1) * 16][32], 72);
      v8f s0f = vzero8(), s1f = vzero8();
      s0f = wmma_f16(qa0, b00, s0f);
      s1f = wmma_f16(qa0, b10, s1f);
      s0f = wmma_f16(qa1, b01, s0f);
      s1f = wmma_f16(qa1, b11, s1f);
#pragma unroll
      for (int r = 0; r < 8; ++r) {
        lP[w][r + 8 * half][16 * (2 * tp) + nl] =
            (_Float16)fast_exp2(s0f[r] - mshift);
        lP[w][r + 8 * half][16 * (2 * tp + 1) + nl] =
            (_Float16)fast_exp2(s1f[r] - mshift);
      }
    }

    v16h pa[4];
#pragma unroll
    for (int s = 0; s < 4; ++s) pa[s] = load_a(&lP[w][0][32 * s], 136);

    // Row sums of P via the matrix pipe: rs = P(16x128) @ ones(128x16).
    v8f rs = vzero8();
#pragma unroll
    for (int s = 0; s < 4; ++s) rs = wmma_f16(pa[s], ones, rs);
#pragma unroll
    for (int r = 0; r < 8; ++r) lrow[r] += rs[r];

    // acc += P(16x128) @ V(128x64): 4 HD tiles x 4 k-steps; preload all four
    // V fragments per HD tile so ds_loads overlap the WMMA pipe.
#pragma unroll
    for (int t = 0; t < 4; ++t) {
      v16h vb[4];
#pragma unroll
      for (int s = 0; s < 4; ++s) vb[s] = load_bt(&lVt[16 * t][32 * s], 136);
#pragma unroll
      for (int s = 0; s < 4; ++s) acc[t] = wmma_f16(pa[s], vb[s], acc[t]);
    }
    __syncthreads();
  }

  // Epilogue: divide by row sum, store f16 into [B*S, D] (head-major cols).
#pragma unroll
  for (int r = 0; r < 8; ++r) {
    float inv = 1.f / lrow[r];
#pragma unroll
    for (int t = 0; t < 4; ++t) {
      int grow = b * S + s0 + r + 8 * half;
      oh[(size_t)grow * D + h * ATTN_HD + 16 * t + nl] =
          (_Float16)(acc[t][r] * inv);
    }
  }
}

// ---------------------------------------------------------------------------
extern "C" void kernel_launch(void* const* d_in, const int* in_sizes, int n_in,
                              void* d_out, int out_size, void* d_ws, size_t ws_size,
                              hipStream_t stream) {
  const float* x     = (const float*)d_in[0];
  const float* ctx   = (const float*)d_in[1];
  const float* Wq    = (const float*)d_in[2];
  const float* bq    = (const float*)d_in[3];
  const float* Wkv   = (const float*)d_in[4];
  const float* bkv   = (const float*)d_in[5];
  const float* Wproj = (const float*)d_in[6];
  const float* bproj = (const float*)d_in[7];
  const float* gq    = (const float*)d_in[8];
  const float* gk    = (const float*)d_in[9];
  float* out = (float*)d_out;

  const int Bb = 4, S = 2048, Sc = 2048, Dd = 1024;
  const size_t elems = (size_t)Bb * S * Dd;   // 8,388,608 per f16 buffer (16MB)

  _Float16* qh = (_Float16*)d_ws;             // [B*S, D]  q * gq * 0.125*log2e
  _Float16* kh = qh + elems;                  // [B,H,Sc,HD]   (post-RMSNorm k)
  _Float16* vh = kh + elems;                  // [B,H,HD,Sc]   (pre-transposed)
  _Float16* oh = vh + elems;                  // [B*S, D]      (attention out)
  float* bound = (float*)(oh + elems);        // scalar log2-score bound

  // 1/sqrt(HD) * log2(e): softmax runs in base 2 downstream.
  const float qscale = 0.125f * 1.44269504088896340736f;

  score_bound_kernel<<<dim3(1), dim3(32), 0, stream>>>(gq, gk, bound);

  // q = rms_norm(x @ Wq + bq, gq) * qscale  -> f16
  gemm_epi_kernel<MODE_Q, float>
      <<<dim3(Dd / 128, (Bb * S) / 128), dim3(256), 0, stream>>>(
          x, Wq, bq, gq, qh, nullptr, nullptr, Bb * S, Dd, Dd, Sc, qscale);

  // kv = ctx @ Wkv + bkv; k = rms_norm(kv[...,0], gk), v = kv[...,1]  -> f16
  gemm_epi_kernel<MODE_KV, float>
      <<<dim3((2 * Dd) / 128, (Bb * Sc) / 128), dim3(256), 0, stream>>>(
          ctx, Wkv, bkv, gk, kh, vh, nullptr, Bb * Sc, 2 * Dd, Dd, Sc, 1.0f);

  // softmax2(q k^T) v  -> f16
  flash_attn_kernel<<<dim3(S / 64, ATTN_H, Bb), dim3(128), 0, stream>>>(
      qh, kh, vh, oh, bound);

  // out = attn_out @ Wproj + bproj  -> f32
  gemm_epi_kernel<MODE_PROJ, _Float16>
      <<<dim3(Dd / 128, (Bb * S) / 128), dim3(256), 0, stream>>>(
          oh, Wproj, bproj, nullptr, nullptr, nullptr, out, Bb * S, Dd, Dd, Sc,
          1.0f);
}
